// TwistedHelicalConduit_21835613733425
// MI455X (gfx1250) — matrix operand order repaired and checked by reading, not compiled, not run on hardware
//
#include <hip/hip_runtime.h>

#define ALPHA   0.35f
#define N_CUBES 131072
#define EMBED   384
#define BATCH   1024
#define TOPK    16

#define BM        128                       // query rows per WG (8 waves x 16)
#define N_CHUNKS  64
#define CHUNK     (N_CUBES / N_CHUNKS)      // 2048 cube columns per WG
#define NITER     (CHUNK / 16)              // 128 x 16-column B tiles
#define KH        392                       // padded f16 row stride (halfs); 196 dw % 64 = 4
#define SSTR      20                        // score-stage row stride (dwords)

typedef float    v2f  __attribute__((ext_vector_type(2)));
typedef float    v8f  __attribute__((ext_vector_type(8)));
typedef _Float16 v4h  __attribute__((ext_vector_type(4)));
typedef _Float16 v8h  __attribute__((ext_vector_type(8)));
typedef _Float16 v16h __attribute__((ext_vector_type(16)));

// ---------------------------------------------------------------------------
// CDNA5 async global->LDS copy (ASYNCcnt-tracked), 16 B per lane.
// ---------------------------------------------------------------------------
__device__ __forceinline__ void async_copy16(unsigned lds_off, const void* gptr) {
    asm volatile("global_load_async_to_lds_b128 %0, %1, off"
                 :: "v"(lds_off), "v"((unsigned long long)(uintptr_t)gptr)
                 : "memory");
}
__device__ __forceinline__ void wait_async0() {
    asm volatile("s_wait_asynccnt 0x0" ::: "memory");
}

// ---------------------------------------------------------------------------
// Sorted top-16 insertion (ascending, slot 0 = min); static indexing only.
// ---------------------------------------------------------------------------
__device__ __forceinline__ void topk_insert(float (&tv)[TOPK], int (&ti)[TOPK],
                                            float v, int vid) {
    if (v > tv[0]) {
#pragma unroll
        for (int i = 0; i < TOPK; ++i) {
            bool gt_next = (i < TOPK - 1) ? (v > tv[i + 1]) : false;
            float nv = gt_next ? tv[i + 1] : v;
            int   ni = gt_next ? ti[i + 1] : vid;
            bool  wr = (v > tv[i]);
            tv[i] = wr ? nv : tv[i];
            ti[i] = wr ? ni : ti[i];
        }
    }
}

// ---------------------------------------------------------------------------
// Kernel 0: L2-normalize queries. One block (384 threads) per row.
// ---------------------------------------------------------------------------
__global__ void norm_q_kernel(const float* __restrict__ q, float* __restrict__ qn) {
    __shared__ float red[EMBED];
    const int row = blockIdx.x;
    const int t   = threadIdx.x;
    const float v = q[row * EMBED + t];
    red[t] = v * v;
    __syncthreads();
    for (int s = 192; s > 2; s >>= 1) {     // 192,96,48,24,12,6,3
        if (t < s) red[t] += red[t + s];
        __syncthreads();
    }
    if (t == 0) {
        float sum = red[0] + red[1] + red[2];
        red[0] = 1.0f / fmaxf(sqrtf(sum), 1e-12f);
    }
    __syncthreads();
    qn[row * EMBED + t] = v * red[0];
}

// ---------------------------------------------------------------------------
// Kernel 1: dual-cone scores. WG tile = 128 queries x 2048 cubes.
//   - f32 read once from HBM; converted on-chip to f16; WMMA f32_16x16x32_f16.
//   - Q tile f16 resident in LDS; E/D tiles async-staged (f32) then converted
//     into double-buffered f16 tiles shared by all 8 waves.
//   - Per-wave register top-16 per query row; partials to workspace.
// ---------------------------------------------------------------------------
__global__ __launch_bounds__(256)
void score_topk_partial_kernel(const float* __restrict__ qn,
                               const float* __restrict__ emb,
                               const float* __restrict__ dual,
                               float* __restrict__ cand_val,
                               int*   __restrict__ cand_idx) {
    __shared__ __align__(16) _Float16 qh[BM * KH];        // 100.4 KB f16 Q tile
    __shared__ __align__(16) _Float16 ebh[2][16 * KH];    // 25.1 KB
    __shared__ __align__(16) _Float16 dbh[2][16 * KH];    // 25.1 KB
    __shared__ __align__(16) float    estg[16 * EMBED];   // 24.6 KB f32 stage
    __shared__ __align__(16) float    dstg[16 * EMBED];   // 24.6 KB f32 stage
    __shared__ float ssc[8 * 16 * SSTR];                  // 10.2 KB

    const int mtile = blockIdx.x;            // 0..7
    const int chunk = blockIdx.y;            // 0..63
    const int t     = threadIdx.x;
    const int wave  = t >> 5;
    const int lane  = t & 31;

    const unsigned lds_es = (unsigned)(uintptr_t)&estg[0];
    const unsigned lds_ds = (unsigned)(uintptr_t)&dstg[0];
    const long chunk_row0 = (long)chunk * CHUNK;

    // Async fill of the f32 stage for tile `it` (rows 2w,2w+1 per wave).
    auto fill_stage = [&](int it) {
        const long rowbase = chunk_row0 + (long)it * 16;
#pragma unroll
        for (int j = 0; j < 6; ++j) {
            const int c   = j * 32 + lane;              // 0..191
            const int row = 2 * wave + (c >= 96 ? 1 : 0);
            const int off = (c % 96) * 16;              // byte offset in row
            const long gb = (rowbase + row) * (long)(EMBED * 4) + off;
            async_copy16(lds_es + row * (EMBED * 4) + off, (const char*)emb  + gb);
            async_copy16(lds_ds + row * (EMBED * 4) + off, (const char*)dual + gb);
        }
    };

    // Cooperative f32 stage -> padded f16 tile conversion.
    auto convert_tile = [&](int buf) {
        _Float16* eh = &ebh[buf][0];
        _Float16* dh = &dbh[buf][0];
#pragma unroll
        for (int j = 0; j < 6; ++j) {
            const int c  = j * 256 + t;                 // 0..1535 float4 chunks
            const int r  = c / 96;
            const int k4 = c - r * 96;
            float4 ve = *(const float4*)(estg + r * EMBED + k4 * 4);
            float4 vd = *(const float4*)(dstg + r * EMBED + k4 * 4);
            v4h he = { (_Float16)ve.x, (_Float16)ve.y, (_Float16)ve.z, (_Float16)ve.w };
            v4h hd = { (_Float16)vd.x, (_Float16)vd.y, (_Float16)vd.z, (_Float16)vd.w };
            *(v4h*)(eh + r * KH + k4 * 4) = he;
            *(v4h*)(dh + r * KH + k4 * 4) = hd;
        }
    };

    // ---- Prologue: stage tile 0 while converting the Q tile to f16 LDS. ----
    fill_stage(0);
    {
        const float4* qsrc = (const float4*)(qn + (size_t)(mtile * BM) * EMBED);
        for (int f = t; f < BM * (EMBED / 4); f += 256) {
            const int m  = f / (EMBED / 4);
            const int k4 = f - m * (EMBED / 4);
            float4 v = qsrc[f];
            v4h h = { (_Float16)v.x, (_Float16)v.y, (_Float16)v.z, (_Float16)v.w };
            *(v4h*)(qh + m * KH + k4 * 4) = h;
        }
    }
    wait_async0();
    __syncthreads();
    convert_tile(0);
    __syncthreads();
    if (NITER > 1) fill_stage(1);

    // ---- Per-row top-16 state (lanes 0..15 own rows wave*16+lane). ----
    float topv[TOPK];
    int   topi[TOPK];
#pragma unroll
    for (int i = 0; i < TOPK; ++i) { topv[i] = -1e30f; topi[i] = 0; }

    const int mloc   = lane & 15;            // A: M row / B: N col for this lane
    const int abase8 = (lane >> 4) << 4;     // A frag: +16 halfs for lanes 16-31
    const int bshft  = (lane >> 4) << 4;     // B frag: +16 halfs for lanes 16-31
    const int mhi    = (lane >> 4) << 3;     // C/D layout row offset
    const _Float16* __restrict__ aRow = qh + (wave * 16 + mloc) * KH + abase8;

    for (int it = 0; it < NITER; ++it) {
        const int buf = it & 1;
        const _Float16* __restrict__ eR = &ebh[buf][0] + mloc * KH + bshft;
        const _Float16* __restrict__ dR = &dbh[buf][0] + mloc * KH + bshft;

        v8f accp = {};
        v8f accd = {};
#pragma unroll
        for (int k0 = 0; k0 < EMBED; k0 += 32) {
            v8h alo = *(const v8h*)(aRow + k0);        // conflict-minimal b128s
            v8h ahi = *(const v8h*)(aRow + k0 + 8);
            v16h a = __builtin_shufflevector(alo, ahi,
                     0,1,2,3,4,5,6,7,8,9,10,11,12,13,14,15);
            v8h belo = *(const v8h*)(eR + k0);
            v8h behi = *(const v8h*)(eR + k0 + 8);
            v16h be = __builtin_shufflevector(belo, behi,
                     0,1,2,3,4,5,6,7,8,9,10,11,12,13,14,15);
            v8h bdlo = *(const v8h*)(dR + k0);
            v8h bdhi = *(const v8h*)(dR + k0 + 8);
            v16h bd = __builtin_shufflevector(bdlo, bdhi,
                     0,1,2,3,4,5,6,7,8,9,10,11,12,13,14,15);
            accp = __builtin_amdgcn_wmma_f32_16x16x32_f16(
                false, a, false, be, (short)0, accp, false, false);
            accd = __builtin_amdgcn_wmma_f32_16x16x32_f16(
                false, a, false, bd, (short)0, accd, false, false);
        }

        // Fused score, staged into this wave's private LDS area.
        float* sb = ssc + wave * (16 * SSTR);
#pragma unroll
        for (int r = 0; r < 8; ++r) {
            float s = accp[r] + ALPHA * fmaxf(accd[r], 0.0f);
            sb[(r + mhi) * SSTR + mloc] = s;
        }
        __builtin_amdgcn_wave_barrier();     // same-wave DS is in-order

        if (lane < 16) {
            const int nbase = chunk * CHUNK + it * 16;
            const float* srow = sb + lane * SSTR;
#pragma unroll
            for (int c = 0; c < 16; ++c) {
                topk_insert(topv, topi, srow[c], nbase + c);
            }
        }

        wait_async0();         // stage now holds tile it+1 (own portion)
        __syncthreads();       // all portions landed; all readers done with buf
        if (it + 1 < NITER) {
            convert_tile(buf ^ 1);              // f16 tile for next iteration
            __syncthreads();                    // conversion visible to all
            if (it + 2 < NITER) fill_stage(it + 2);   // overlaps next compute
        }
    }

    if (lane < 16) {
        const int q = mtile * BM + wave * 16 + lane;
        float* cv = cand_val + ((size_t)chunk * BATCH + q) * TOPK;
        int*   ci = cand_idx + ((size_t)chunk * BATCH + q) * TOPK;
#pragma unroll
        for (int i = 0; i < TOPK; ++i) { cv[i] = topv[i]; ci[i] = topi[i]; }
    }
}

// ---------------------------------------------------------------------------
// Kernel 2: merge 64 chunk-partials per query -> final descending top-16.
// ---------------------------------------------------------------------------
__global__ void final_topk_kernel(const float* __restrict__ cand_val,
                                  const int*   __restrict__ cand_idx,
                                  float* __restrict__ out_val,
                                  int*   __restrict__ out_idx) {
    const int q = blockIdx.x * blockDim.x + threadIdx.x;
    if (q >= BATCH) return;

    float topv[TOPK];
    int   topi[TOPK];
#pragma unroll
    for (int i = 0; i < TOPK; ++i) { topv[i] = -1e30f; topi[i] = 0; }

    for (int chunk = 0; chunk < N_CHUNKS; ++chunk) {
        const float* cv = cand_val + ((size_t)chunk * BATCH + q) * TOPK;
        const int*   ci = cand_idx + ((size_t)chunk * BATCH + q) * TOPK;
#pragma unroll
        for (int j = 0; j < TOPK; ++j) {
            topk_insert(topv, topi, cv[j], ci[j]);
        }
    }
#pragma unroll
    for (int j = 0; j < TOPK; ++j) {
        out_val[q * TOPK + j] = topv[TOPK - 1 - j];
        out_idx[q * TOPK + j] = topi[TOPK - 1 - j];
    }
}

// ---------------------------------------------------------------------------
extern "C" void kernel_launch(void* const* d_in, const int* in_sizes, int n_in,
                              void* d_out, int out_size, void* d_ws, size_t ws_size,
                              hipStream_t stream) {
    const float* q    = (const float*)d_in[0];   // [1024, 384]
    const float* emb  = (const float*)d_in[1];   // [131072, 384]
    const float* dual = (const float*)d_in[2];   // [131072, 384]
    // d_in[3] = top_k scalar (fixed at 16)

    float* qn       = (float*)d_ws;                                  // 1.5 MB
    float* cand_val = qn + (size_t)BATCH * EMBED;                    // 4 MB
    int*   cand_idx = (int*)(cand_val + (size_t)N_CHUNKS * BATCH * TOPK); // 4 MB

    float* out_val = (float*)d_out;                  // [1024,16] f32
    int*   out_idx = (int*)d_out + BATCH * TOPK;     // [1024,16] i32

    norm_q_kernel<<<BATCH, EMBED, 0, stream>>>(q, qn);

    dim3 grid(BATCH / BM, N_CHUNKS);                 // (8 mtiles, 64 chunks)
    score_topk_partial_kernel<<<grid, 256, 0, stream>>>(qn, emb, dual,
                                                        cand_val, cand_idx);

    final_topk_kernel<<<(BATCH + 255) / 256, 256, 0, stream>>>(cand_val, cand_idx,
                                                               out_val, out_idx);
}
